// IxformerGroupQuantGemmCombineMoE_80908593922369
// MI455X (gfx1250) — 8-bit matrix operand order repaired and checked by reading, compiled, not run on hardware
//
#include <hip/hip_runtime.h>
#include <stdint.h>

#define T_TOK   16384
#define TOPK    8
#define KDIM    512
#define NDIM    512
#define NEXP    32
#define ROWS    (T_TOK * TOPK)
#define MT      64
#define NT      64

typedef int   v8i __attribute__((ext_vector_type(8)));
typedef float v8f __attribute__((ext_vector_type(8)));

__device__ __forceinline__ float bf16_round(float x) {
    unsigned u = __float_as_uint(x);
    u = (u + 0x7FFFu + ((u >> 16) & 1u)) & 0xFFFF0000u;
    return __uint_as_float(u);
}
__device__ __forceinline__ unsigned short f32_to_bf16(float x) {
    unsigned u = __float_as_uint(x);
    u = u + 0x7FFFu + ((u >> 16) & 1u);
    return (unsigned short)(u >> 16);
}
__device__ __forceinline__ float bf16_to_f32(unsigned short h) {
    return __uint_as_float(((unsigned)h) << 16);
}

__global__ __launch_bounds__(128)
void IxformerGroupQuantGemmCombineMoE_kernel(
        const int8_t*  __restrict__ inp,
        const int8_t*  __restrict__ wgt,
        const float*   __restrict__ gates,
        const int*     __restrict__ tok_idx,
        const int*     __restrict__ src2dst,
        const int*     __restrict__ tok_cnt,
        const unsigned short* __restrict__ shared_out,
        const float*   __restrict__ wscale,
        const float*   __restrict__ iscale,
        unsigned short* __restrict__ out)
{
    __shared__ __align__(16) unsigned char sA[MT * KDIM];
    __shared__ __align__(16) unsigned char sB[NT * KDIM];
    __shared__ int   sSrc[MT];
    __shared__ float sIS[MT * 4];
    __shared__ float sWS[4];

    const int tid  = threadIdx.x;
    const int lane = tid & 31;
    const int wave = tid >> 5;
    const int R    = blockIdx.x * MT;
    const int n0   = blockIdx.y * NT;
    const int nb   = n0 >> 7;

    int e = 0, c = 0;
    for (int i = 0; i < NEXP - 1; ++i) {
        int nxt = c + tok_cnt[i];
        if (R >= nxt) { c = nxt; e = i + 1; }
    }

    if (tid < MT) sSrc[tid] = tok_idx[R + tid];
    __syncthreads();

    {
        #pragma unroll
        for (int i = 0; i < (MT * KDIM / 16) / 128; ++i) {
            int id  = tid + i * 128;
            int row = id >> 5;
            int off = (id & 31) * 16;
            unsigned long long ga = (unsigned long long)(inp + (size_t)sSrc[row] * KDIM + off);
            unsigned la = (unsigned)(size_t)(&sA[row * KDIM + off]);
            asm volatile("global_load_async_to_lds_b128 %0, %1, off"
                         :: "v"(la), "v"(ga) : "memory");
        }
    }
    #pragma unroll
    for (int i = 0; i < 2; ++i) {
        int idx = tid + i * 128;
        int row = idx >> 2, kb = idx & 3;
        sIS[idx] = iscale[(size_t)sSrc[row] * 4 + kb];
    }
    if (tid < 4) sWS[tid] = wscale[((size_t)e * 4 + tid) * 4 + nb];

    {
        const size_t eOff = (size_t)e * KDIM * NDIM;
        int ncg = tid & 3;
        int kb0 = (tid >> 2) * 16;
        for (int g = 0; g < 4; ++g) {
            int k4 = kb0 + g * 4;
            const int8_t* gb = wgt + eOff + (size_t)k4 * NDIM + n0 + ncg * 16;
            uint4 r0 = *(const uint4*)(gb);
            uint4 r1 = *(const uint4*)(gb + NDIM);
            uint4 r2 = *(const uint4*)(gb + 2 * NDIM);
            uint4 r3 = *(const uint4*)(gb + 3 * NDIM);
            const unsigned* w0 = (const unsigned*)&r0;
            const unsigned* w1 = (const unsigned*)&r1;
            const unsigned* w2 = (const unsigned*)&r2;
            const unsigned* w3 = (const unsigned*)&r3;
            #pragma unroll
            for (int ws_ = 0; ws_ < 4; ++ws_) {
                #pragma unroll
                for (int p = 0; p < 4; ++p) {
                    unsigned sel01 = ((4u + (unsigned)p) << 8) | (unsigned)p;
                    unsigned t01 = __builtin_amdgcn_perm(w1[ws_], w0[ws_], sel01);
                    unsigned t23 = __builtin_amdgcn_perm(w3[ws_], w2[ws_], sel01);
                    unsigned d   = __builtin_amdgcn_perm(t23, t01, 0x05040100u);
                    *(unsigned*)(sB + (ncg * 16 + ws_ * 4 + p) * KDIM + k4) = d;
                }
            }
        }
    }
    asm volatile("s_wait_asynccnt 0x0" ::: "memory");
    __syncthreads();

    const int half = lane >> 4;
    const int lm   = lane & 15;
    const int mrow = wave * 16 + lm;

    v8f C[4];
    #pragma unroll
    for (int ns = 0; ns < 4; ++ns) C[ns] = (v8f){0.f,0.f,0.f,0.f,0.f,0.f,0.f,0.f};

    for (int kb = 0; kb < 4; ++kb) {
        v8i I[4];
        #pragma unroll
        for (int ns = 0; ns < 4; ++ns) I[ns] = (v8i){0,0,0,0,0,0,0,0};

        #pragma unroll
        for (int h = 0; h < 2; ++h) {
            int k0 = kb * 128 + h * 64;
            const unsigned* ap = (const unsigned*)(sA + mrow * KDIM + k0 + half * 8);
            v8i A;
            A[0] = (int)ap[0];  A[1] = (int)ap[1];
            A[2] = (int)ap[4];  A[3] = (int)ap[5];
            A[4] = (int)ap[8];  A[5] = (int)ap[9];
            A[6] = (int)ap[12]; A[7] = (int)ap[13];
            #pragma unroll
            for (int ns = 0; ns < 4; ++ns) {
                const unsigned* bp = (const unsigned*)(sB + (ns * 16 + lm) * KDIM + k0 + half * 8);
                v8i Bm;
                Bm[0] = (int)bp[0]; Bm[1] = (int)bp[1]; Bm[2] = (int)bp[4];  Bm[3] = (int)bp[5];
                Bm[4] = (int)bp[8]; Bm[5] = (int)bp[9]; Bm[6] = (int)bp[12]; Bm[7] = (int)bp[13];
                I[ns] = __builtin_amdgcn_wmma_i32_16x16x64_iu8(true, A, true, Bm, I[ns], false, false);
            }
        }
        float ws = sWS[kb];
        float sc[8];
        #pragma unroll
        for (int j = 0; j < 8; ++j)
            sc[j] = sIS[(wave * 16 + half * 8 + j) * 4 + kb] * ws;
        #pragma unroll
        for (int ns = 0; ns < 4; ++ns)
            #pragma unroll
            for (int j = 0; j < 8; ++j)
                C[ns][j] += (float)I[ns][j] * sc[j];
    }

    const int tTok = blockIdx.x * (MT / TOPK) + wave * 2 + half;
    float g[8]; float gsum = 0.f;
    #pragma unroll
    for (int j = 0; j < 8; ++j) { g[j] = gates[(size_t)tTok * TOPK + j]; gsum += g[j]; }
    float inv = 1.f / fmaxf(gsum, 1e-12f);
    float gg[8];
    #pragma unroll
    for (int j = 0; j < 8; ++j) {
        int sd = src2dst[(size_t)tTok * TOPK + j];
        gg[j] = (sd == -1) ? 0.f : g[j] * inv;
    }
    #pragma unroll
    for (int ns = 0; ns < 4; ++ns) {
        float acc = 0.f;
        #pragma unroll
        for (int j = 0; j < 8; ++j)
            acc += gg[j] * bf16_round(C[ns][j]);
        int ncol = n0 + ns * 16 + lm;
        acc += bf16_to_f32(shared_out[(size_t)tTok * NDIM + ncol]);
        out[(size_t)tTok * NDIM + ncol] = f32_to_bf16(acc);
    }
}

extern "C" void kernel_launch(void* const* d_in, const int* in_sizes, int n_in,
                              void* d_out, int out_size, void* d_ws, size_t ws_size,
                              hipStream_t stream) {
    (void)in_sizes; (void)n_in; (void)out_size; (void)d_ws; (void)ws_size;
    const int8_t*  inp      = (const int8_t*)d_in[0];
    const int8_t*  wgt      = (const int8_t*)d_in[1];
    const float*   gates    = (const float*)d_in[2];
    const int*     tok_idx  = (const int*)d_in[3];
    const int*     src2dst  = (const int*)d_in[4];
    const int*     tok_cnt  = (const int*)d_in[5];
    const unsigned short* sh = (const unsigned short*)d_in[6];
    const float*   wscale   = (const float*)d_in[7];
    const float*   iscale   = (const float*)d_in[8];
    unsigned short* out     = (unsigned short*)d_out;

    dim3 grid(ROWS / MT, NDIM / NT, 1);
    dim3 block(128, 1, 1);
    IxformerGroupQuantGemmCombineMoE_kernel<<<grid, block, 0, stream>>>(
        inp, wgt, gates, tok_idx, src2dst, tok_cnt, sh, wscale, iscale, out);
}
